// QwenAttention_11768210391188
// MI455X (gfx1250) — compile-verified, matrix operand order checked by reading, not run on hardware
//
#include <hip/hip_runtime.h>

typedef unsigned short u16t;
typedef unsigned int   u32t;
typedef unsigned long long u64t;
typedef __attribute__((ext_vector_type(16))) __bf16 bf16x16;
typedef __attribute__((ext_vector_type(8)))  float  f32x8;

union FragB { bf16x16 v; uint4 u[2]; };

__device__ __forceinline__ u16t f2bf(float f) {
    u32t u = __float_as_uint(f);
    u += 0x7FFFu + ((u >> 16) & 1u);   // round-to-nearest-even
    return (u16t)(u >> 16);
}
__device__ __forceinline__ float bf2f(u16t h) {
    return __uint_as_float(((u32t)h) << 16);
}

// gfx1250 async global->LDS copy (ASYNCcnt-tracked), 16 B per lane.
// lds_off: wave-relative LDS byte offset; voff: per-lane byte offset from base.
__device__ __forceinline__ void async_load_b128(u32t lds_off, const void* gbase, u32t voff) {
    asm volatile("global_load_async_to_lds_b128 %0, %1, %2"
                 :: "v"(lds_off), "v"(voff), "s"((u64t)gbase)
                 : "memory");
}
__device__ __forceinline__ void wait_async0() {
    asm volatile("s_wait_asynccnt 0x0" ::: "memory");
}
__device__ __forceinline__ u32t lds_addr(const void* p) {
    return (u32t)(size_t)p;   // low 32 bits of flat LDS address = LDS offset
}

// ---------------------------------------------------------------------------
// fp32 -> bf16 elementwise convert (vectorized x4)
// ---------------------------------------------------------------------------
__global__ __launch_bounds__(256)
void cvt_kernel(const float* __restrict__ in, u16t* __restrict__ out, int n4) {
    int i = blockIdx.x * 256 + threadIdx.x;
    if (i < n4) {
        float4 f = ((const float4*)in)[i];
        u32t lo = (u32t)f2bf(f.x) | ((u32t)f2bf(f.y) << 16);
        u32t hi = (u32t)f2bf(f.z) | ((u32t)f2bf(f.w) << 16);
        ((u32t*)out)[2 * i]     = lo;
        ((u32t*)out)[2 * i + 1] = hi;
    }
}

// ---------------------------------------------------------------------------
// fp32 (R x C) -> bf16 transposed (C x R), 32x32 LDS tiles
// ---------------------------------------------------------------------------
__global__ __launch_bounds__(256)
void trans_kernel(const float* __restrict__ in, u16t* __restrict__ out, int R, int C) {
    __shared__ float tile[32][33];
    int tx = threadIdx.x & 31, ty = threadIdx.x >> 5;
    int c = blockIdx.x * 32 + tx;
#pragma unroll
    for (int i = 0; i < 4; ++i) {
        int r = blockIdx.y * 32 + ty + i * 8;
        tile[ty + i * 8][tx] = in[(size_t)r * C + c];
    }
    __syncthreads();
#pragma unroll
    for (int i = 0; i < 4; ++i) {
        int oc = blockIdx.x * 32 + ty + i * 8;           // output row = input col
        out[(size_t)oc * R + blockIdx.y * 32 + tx] = f2bf(tile[tx][ty + i * 8]);
    }
}

// ---------------------------------------------------------------------------
// bf16 GEMM: C[M x N] = A[M x K] * Bt[N x K]^T (+ bias), f32 accumulate.
// Block tile 128x128, 8 waves of 32x64. K staged 64/iter, double-buffered LDS
// fed by global_load_async_to_lds_b128. One barrier per iteration.
// LDS row stride 72 bf16 (144 B): frag b128 loads hit 16 distinct banks.
// ---------------------------------------------------------------------------
template <bool OUT_F32>
__global__ __launch_bounds__(256)
void gemm_bf16_kernel(const u16t* __restrict__ A, const u16t* __restrict__ Bt,
                      const float* __restrict__ bias, void* __restrict__ Cout,
                      int M, int N, int K) {
    __shared__ __align__(16) u16t lds_a[2][128 * 72];
    __shared__ __align__(16) u16t lds_b[2][128 * 72];
    const int tid = threadIdx.x;
    const int lane = tid & 31, wave = tid >> 5;
    const int wm = wave >> 1, wn = wave & 1;
    const int m0 = blockIdx.x * 128, n0 = blockIdx.y * 128;
    const int lr = lane & 15, lh = (lane < 16) ? 0 : 8;
    const int srow = tid >> 1, shalf = tid & 1;

    // per-thread staging addresses: row = srow, 16 bf16 (two b128, at +0 and +16) from shalf*32
    const u32t a_goff = ((u32t)(m0 + srow) * (u32t)K + shalf * 32) * 2u;
    const u32t b_goff = ((u32t)(n0 + srow) * (u32t)K + shalf * 32) * 2u;
    const u32t l_off  = (u32t)(srow * 72 + shalf * 32) * 2u;

    auto prefetch = [&](int buf, int kk) {
        u32t la = lds_addr(&lds_a[buf][0]) + l_off;
        u32t lb = lds_addr(&lds_b[buf][0]) + l_off;
        u32t ka = a_goff + (u32t)kk * 2u;
        u32t kb = b_goff + (u32t)kk * 2u;
        async_load_b128(la,      A,  ka);
        async_load_b128(la + 32, A,  ka + 32);
        async_load_b128(lb,      Bt, kb);
        async_load_b128(lb + 32, Bt, kb + 32);
    };

    f32x8 acc[2][4] = {};
    const int niter = K / 64;
    prefetch(0, 0);

    for (int it = 0; it < niter; ++it) {
        const int buf = it & 1;
        wait_async0();           // my async loads into lds[buf] are done
        __syncthreads();         // everyone's are done; lds[buf^1] free to reuse
        if (it + 1 < niter) prefetch(buf ^ 1, (it + 1) * 64);

        const u16t* la = lds_a[buf];
        const u16t* lb = lds_b[buf];
        FragB af[2][2];
#pragma unroll
        for (int ms = 0; ms < 2; ++ms)
#pragma unroll
            for (int kb = 0; kb < 2; ++kb) {
                int r = wm * 32 + ms * 16 + lr;
                af[ms][kb].u[0] = *(const uint4*)&la[r * 72 + kb * 32 + lh];
                af[ms][kb].u[1] = *(const uint4*)&la[r * 72 + kb * 32 + lh + 16];
            }
#pragma unroll
        for (int ns = 0; ns < 4; ++ns) {
#pragma unroll
            for (int kb = 0; kb < 2; ++kb) {
                FragB bfv;
                int r = wn * 64 + ns * 16 + lr;
                bfv.u[0] = *(const uint4*)&lb[r * 72 + kb * 32 + lh];
                bfv.u[1] = *(const uint4*)&lb[r * 72 + kb * 32 + lh + 16];
#pragma unroll
                for (int ms = 0; ms < 2; ++ms)
                    acc[ms][ns] = __builtin_amdgcn_wmma_f32_16x16x32_bf16(
                        false, af[ms][kb].v, false, bfv.v, (short)0, acc[ms][ns], false, false);
            }
        }
    }

    // epilogue: C tile layout — lane holds column N=lr, rows j + 8*(lane>=16)
#pragma unroll
    for (int ms = 0; ms < 2; ++ms) {
#pragma unroll
        for (int ns = 0; ns < 4; ++ns) {
            int n = n0 + wn * 64 + ns * 16 + lr;
            float bv = bias ? bias[n] : 0.0f;
            int mbase = m0 + wm * 32 + ms * 16 + ((lane < 16) ? 0 : 8);
#pragma unroll
            for (int j = 0; j < 8; ++j) {
                float v = acc[ms][ns][j] + bv;
                size_t off = (size_t)(mbase + j) * N + n;
                if (OUT_F32) ((float*)Cout)[off] = v;
                else         ((u16t*)Cout)[off] = f2bf(v);
            }
        }
    }
}

// ---------------------------------------------------------------------------
// RoPE (in-place on Q slice of QKV), K -> (b,hk,s,d), V -> (b,hk,d,s) transposed
// ---------------------------------------------------------------------------
__global__ __launch_bounds__(128)
void rope_kernel(const int* __restrict__ positions, u16t* __restrict__ qkv,
                 u16t* __restrict__ Kb, u16t* __restrict__ Vt) {
    const int t = blockIdx.x;
    const int b = t >> 11, s = t & 2047;
    const float pos = (float)positions[t];
    u16t* base = qkv + (size_t)t * 4608;
    const int tid = threadIdx.x;

    for (int idx = tid; idx < 28 * 64; idx += 128) {            // Q: rotate half-pairs
        int h = idx >> 6, d = idx & 63;
        float inv = __powf(1000000.0f, -(float)d / 64.0f);
        float sn, cs; __sincosf(pos * inv, &sn, &cs);
        float x1 = bf2f(base[h * 128 + d]);
        float x2 = bf2f(base[h * 128 + 64 + d]);
        base[h * 128 + d]      = f2bf(x1 * cs - x2 * sn);
        base[h * 128 + 64 + d] = f2bf(x2 * cs + x1 * sn);
    }
    for (int idx = tid; idx < 4 * 64; idx += 128) {             // K: rotate + relayout
        int hk = idx >> 6, d = idx & 63;
        float inv = __powf(1000000.0f, -(float)d / 64.0f);
        float sn, cs; __sincosf(pos * inv, &sn, &cs);
        float x1 = bf2f(base[3584 + hk * 128 + d]);
        float x2 = bf2f(base[3584 + hk * 128 + 64 + d]);
        size_t ko = (size_t)((b * 4 + hk) * 2048 + s) * 128 + d;
        Kb[ko]      = f2bf(x1 * cs - x2 * sn);
        Kb[ko + 64] = f2bf(x2 * cs + x1 * sn);
    }
    for (int idx = tid; idx < 4 * 128; idx += 128) {            // V: transpose copy
        int hk = idx >> 7, d = idx & 127;
        Vt[(size_t)((b * 4 + hk) * 128 + d) * 2048 + s] = base[4096 + hk * 128 + d];
    }
}

// ---------------------------------------------------------------------------
// Flash attention: grid (B*28 heads, S/128). Wave = 16 q rows x 128 d output.
// KV tiles of 64 double-buffered in LDS via async copies; online softmax;
// P relayout (C-form -> A-form) through per-wave LDS scratch.
// ---------------------------------------------------------------------------
__global__ __launch_bounds__(256)
void attn_kernel(const u16t* __restrict__ qkv,   // (4096 x 4608), q roped
                 const u16t* __restrict__ Kb,    // (B*4, 2048, 128)
                 const u16t* __restrict__ Vt,    // (B*4, 128, 2048)
                 u16t* __restrict__ Ob) {        // (4096 x 3584)
    __shared__ __align__(16) u16t lds_k[2][64 * 136];   // kv-row major, pad 8
    __shared__ __align__(16) u16t lds_v[2][128 * 72];   // dim-row major (V^T), pad 8
    __shared__ __align__(16) u16t lds_p[8 * 16 * 72];
    const int tid = threadIdx.x;
    const int lane = tid & 31, wave = tid >> 5;
    const int b = blockIdx.x / 28, h = blockIdx.x % 28, hk = h / 7;
    const int q0 = blockIdx.y * 128;
    const int lr = lane & 15, lh = (lane < 16) ? 0 : 8;
    const int qrow = q0 + wave * 16;
    const float scale = 0.08838834764831845f;   // 1/sqrt(128)

    // staging addresses (per thread): K row r, quarter q4; V^T row dr, half hf
    const int kr = tid >> 2, kq4 = tid & 3;
    const int vr = tid >> 1, vhf = tid & 1;
    const u32t k_goff = ((u32t)((b * 4 + hk) * 2048 + kr) * 128u + kq4 * 32) * 2u;
    const u32t v_goff = ((u32t)((b * 4 + hk) * 128 + vr) * 2048u + vhf * 32) * 2u;
    const u32t k_loff = (u32t)(kr * 136 + kq4 * 32) * 2u;
    const u32t v_loff = (u32t)(vr * 72 + vhf * 32) * 2u;

    auto prefetch_kv = [&](int buf, int kv0) {
        u32t lk = lds_addr(&lds_k[buf][0]) + k_loff;
        u32t lv = lds_addr(&lds_v[buf][0]) + v_loff;
        u32t gk = k_goff + (u32t)kv0 * 256u;   // kv step: *128 elems *2 B
        u32t gv = v_goff + (u32t)kv0 * 2u;     // kv step along row: *2 B
        async_load_b128(lk,      Kb, gk);
        async_load_b128(lk + 32, Kb, gk + 32);
        async_load_b128(lv,      Vt, gv);
        async_load_b128(lv + 32, Vt, gv + 32);
    };

    FragB qf[4];
    {
        const u16t* qp = qkv + (size_t)(b * 2048 + qrow + lr) * 4608 + h * 128;
#pragma unroll
        for (int kb = 0; kb < 4; ++kb) {
            qf[kb].u[0] = *(const uint4*)(qp + kb * 32 + lh);
            qf[kb].u[1] = *(const uint4*)(qp + kb * 32 + lh + 16);
        }
    }

    float m_i[8], l_i[8];
    f32x8 o[8] = {};
#pragma unroll
    for (int j = 0; j < 8; ++j) { m_i[j] = -3.0e38f; l_i[j] = 0.0f; }

    u16t* pbuf = &lds_p[wave * 16 * 72];
    const int ntiles = (q0 + 128) / 64;          // causal bound for this block
    prefetch_kv(0, 0);

    for (int t = 0; t < ntiles; ++t) {
        const int kv0 = t * 64;
        const int buf = t & 1;
        wait_async0();
        __syncthreads();
        if (t + 1 < ntiles) prefetch_kv(buf ^ 1, kv0 + 64);
        if (kv0 > qrow + 15) continue;           // fully masked for this wave

        const u16t* lk = lds_k[buf];
        const u16t* lv = lds_v[buf];

        // ---- scores S = Q K^T (16 x 64) ----
        f32x8 s4[4] = {};
#pragma unroll
        for (int ns = 0; ns < 4; ++ns) {
#pragma unroll
            for (int kb = 0; kb < 4; ++kb) {
                FragB kf;
                int r = ns * 16 + lr;
                kf.u[0] = *(const uint4*)&lk[r * 136 + kb * 32 + lh];
                kf.u[1] = *(const uint4*)&lk[r * 136 + kb * 32 + lh + 16];
                s4[ns] = __builtin_amdgcn_wmma_f32_16x16x32_bf16(
                    false, qf[kb].v, false, kf.v, (short)0, s4[ns], false, false);
            }
        }
        // ---- scale + causal mask + per-row max ----
        float rmax[8];
#pragma unroll
        for (int j = 0; j < 8; ++j) rmax[j] = -3.0e38f;
#pragma unroll
        for (int ns = 0; ns < 4; ++ns) {
            int kvcol = kv0 + ns * 16 + lr;
#pragma unroll
            for (int j = 0; j < 8; ++j) {
                int sq = qrow + j + ((lane < 16) ? 0 : 8);
                float v = s4[ns][j] * scale;
                v = (kvcol <= sq) ? v : -3.0e38f;
                s4[ns][j] = v;
                rmax[j] = fmaxf(rmax[j], v);
            }
        }
#pragma unroll
        for (int j = 0; j < 8; ++j) {
            float v = rmax[j];
            v = fmaxf(v, __shfl_xor(v, 1, 32));
            v = fmaxf(v, __shfl_xor(v, 2, 32));
            v = fmaxf(v, __shfl_xor(v, 4, 32));
            v = fmaxf(v, __shfl_xor(v, 8, 32));
            rmax[j] = v;
        }
        float alpha[8], rsum[8];
#pragma unroll
        for (int j = 0; j < 8; ++j) {
            float mnew = fmaxf(m_i[j], rmax[j]);
            alpha[j] = __expf(m_i[j] - mnew);
            m_i[j] = mnew;
            rsum[j] = 0.0f;
        }
#pragma unroll
        for (int ns = 0; ns < 4; ++ns)
#pragma unroll
            for (int j = 0; j < 8; ++j) {
                float p = __expf(s4[ns][j] - m_i[j]);
                s4[ns][j] = p;
                rsum[j] += p;
            }
#pragma unroll
        for (int j = 0; j < 8; ++j) {
            float r = rsum[j];
            r += __shfl_xor(r, 1, 32);
            r += __shfl_xor(r, 2, 32);
            r += __shfl_xor(r, 4, 32);
            r += __shfl_xor(r, 8, 32);
            l_i[j] = l_i[j] * alpha[j] + r;
        }
#pragma unroll
        for (int ns = 0; ns < 8; ++ns)
#pragma unroll
            for (int j = 0; j < 8; ++j)
                o[ns][j] *= alpha[j];

        // ---- P: C-layout -> A-layout via per-wave LDS ----
#pragma unroll
        for (int ns = 0; ns < 4; ++ns)
#pragma unroll
            for (int j = 0; j < 8; ++j) {
                int m = j + ((lane < 16) ? 0 : 8);
                pbuf[m * 72 + ns * 16 + lr] = f2bf(s4[ns][j]);
            }
        // ---- O += P V ----
#pragma unroll
        for (int kb = 0; kb < 2; ++kb) {
            FragB pf;
            pf.u[0] = *(const uint4*)&pbuf[lr * 72 + kb * 32 + lh];
            pf.u[1] = *(const uint4*)&pbuf[lr * 72 + kb * 32 + lh + 16];
#pragma unroll
            for (int ns = 0; ns < 8; ++ns) {
                FragB vf;
                int r = ns * 16 + lr;
                vf.u[0] = *(const uint4*)&lv[r * 72 + kb * 32 + lh];
                vf.u[1] = *(const uint4*)&lv[r * 72 + kb * 32 + lh + 16];
                o[ns] = __builtin_amdgcn_wmma_f32_16x16x32_bf16(
                    false, pf.v, false, vf.v, (short)0, o[ns], false, false);
            }
        }
    }

    // ---- normalize + write O (bf16) ----
#pragma unroll
    for (int ns = 0; ns < 8; ++ns)
#pragma unroll
        for (int j = 0; j < 8; ++j) {
            int sr = qrow + j + ((lane < 16) ? 0 : 8);
            float v = o[ns][j] / l_i[j];
            Ob[(size_t)(b * 2048 + sr) * 3584 + h * 128 + ns * 16 + lr] = f2bf(v);
        }
}

// ---------------------------------------------------------------------------
extern "C" void kernel_launch(void* const* d_in, const int* in_sizes, int n_in,
                              void* d_out, int out_size, void* d_ws, size_t ws_size,
                              hipStream_t stream) {
    (void)in_sizes; (void)n_in; (void)out_size; (void)ws_size;
    const int*   positions = (const int*)d_in[0];
    const float* X         = (const float*)d_in[1];
    const float* Wqkv      = (const float*)d_in[2];
    const float* bqkv      = (const float*)d_in[3];
    const float* Wo        = (const float*)d_in[4];
    float* out = (float*)d_out;
    char*  ws  = (char*)d_ws;

    // workspace layout (bytes), total = 128 MB
    u16t* Xb    = (u16t*)(ws + 0);           // 4096x3584 bf16 (reused as O later)
    u16t* WqkvT = (u16t*)(ws + 29360128);    // 4608x3584 bf16
    u16t* WoT   = (u16t*)(ws + 62390272);    // 3584x3584 bf16
    u16t* QKV   = (u16t*)(ws + 88080384);    // 4096x4608 bf16
    u16t* Kb    = (u16t*)(ws + 125829120);   // (2*4,2048,128) bf16
    u16t* Vt    = (u16t*)(ws + 130023424);   // (2*4,128,2048) bf16

    cvt_kernel<<<(4096 * 3584 / 4) / 256, 256, 0, stream>>>(X, Xb, 4096 * 3584 / 4);
    trans_kernel<<<dim3(4608 / 32, 3584 / 32), 256, 0, stream>>>(Wqkv, WqkvT, 3584, 4608);
    trans_kernel<<<dim3(3584 / 32, 3584 / 32), 256, 0, stream>>>(Wo, WoT, 3584, 3584);

    gemm_bf16_kernel<false><<<dim3(32, 36), 256, 0, stream>>>(
        Xb, WqkvT, bqkv, QKV, 4096, 4608, 3584);

    rope_kernel<<<4096, 128, 0, stream>>>(positions, QKV, Kb, Vt);

    attn_kernel<<<dim3(56, 16), 256, 0, stream>>>(QKV, Kb, Vt, Xb);

    gemm_bf16_kernel<true><<<dim3(32, 28), 256, 0, stream>>>(
        Xb, WoT, nullptr, out, 4096, 3584, 3584);
}
